// DomainAttention_20899310862964
// MI455X (gfx1250) — matrix operand order repaired and checked by reading, not compiled, not run on hardware
//
#include <hip/hip_runtime.h>
#include <hip/hip_bf16.h>

#define B_DIM 4096
#define D_DIM 768
#define N_DIM 8192
#define S_DIM 4
#define NCH   (N_DIM / S_DIM)   // 2048 rows per source chunk

typedef __attribute__((ext_vector_type(16))) __bf16 v16bf;
typedef __attribute__((ext_vector_type(8)))  float  v8f;

// ---------------------------------------------------------------------------
// Kernel 1: fp32 -> bf16 (round to nearest even)
// ---------------------------------------------------------------------------
__global__ void cvt_bf16_kernel(const float* __restrict__ in,
                                unsigned short* __restrict__ out, int n) {
    int i = blockIdx.x * blockDim.x + threadIdx.x;
    int stride = gridDim.x * blockDim.x;
    for (; i < n; i += stride) {
        unsigned int u = __float_as_uint(in[i]);
        unsigned int r = (u + 0x7FFFu + ((u >> 16) & 1u)) >> 16;
        out[i] = (unsigned short)r;
    }
}

// ---------------------------------------------------------------------------
// Kernel 2: theta[s,b] = exp(batch_x[b,:] . phi[s,:])   (S=4, tiny GEMM)
// one wave per b; lane-strided dot, shfl_xor reduce
// ---------------------------------------------------------------------------
__global__ void theta_kernel(const float* __restrict__ x,
                             const float* __restrict__ phi,
                             float* __restrict__ theta) {
    const int lane = threadIdx.x & 31;
    const int wave = threadIdx.x >> 5;
    const int b = blockIdx.x * (blockDim.x >> 5) + wave;
    if (b >= B_DIM) return;
    float acc0 = 0.f, acc1 = 0.f, acc2 = 0.f, acc3 = 0.f;
    const float* xr = x + (size_t)b * D_DIM;
    for (int d = lane; d < D_DIM; d += 32) {
        float xv = xr[d];
        acc0 += xv * phi[0 * D_DIM + d];
        acc1 += xv * phi[1 * D_DIM + d];
        acc2 += xv * phi[2 * D_DIM + d];
        acc3 += xv * phi[3 * D_DIM + d];
    }
    #pragma unroll
    for (int off = 16; off > 0; off >>= 1) {
        acc0 += __shfl_xor(acc0, off, 32);
        acc1 += __shfl_xor(acc1, off, 32);
        acc2 += __shfl_xor(acc2, off, 32);
        acc3 += __shfl_xor(acc3, off, 32);
    }
    if (lane == 0) {
        theta[0 * B_DIM + b] = __expf(acc0);
        theta[1 * B_DIM + b] = __expf(acc1);
        theta[2 * B_DIM + b] = __expf(acc2);
        theta[3 * B_DIM + b] = __expf(acc3);
    }
}

// ---------------------------------------------------------------------------
// Kernel 3: con[n,b] = cat[n,:] . batch_x[b,:]  via v_wmma_f32_16x16x32_bf16
// Block = 128 threads = 4 waves; block tile 64x64, wave tile 32x32.
// Operands (19 MB bf16 total) live in L2 -> direct global fragment loads.
//
// bf16 A (16x32 MxK) per-lane layout (ISA 7.12.2): lane l holds row m=l&15,
//   elems 0..7  = K kb..kb+7,  elems 8..15 = K kb+16..kb+23,  kb=(l>>4)*8.
// bf16 B (32x16 KxN): lane l holds col n=l&15, elems 0..7 = K kb..kb+7, elems 8..15 = K kb+16..kb+23 (as A).
// C/D f32: VGPR e, lane l -> (M = e + 8*(l>>4), N = l&15).
// ---------------------------------------------------------------------------
__global__ void __launch_bounds__(128)
gemm_con_kernel(const unsigned short* __restrict__ catBF,
                const unsigned short* __restrict__ xBF,
                float* __restrict__ con) {
    const int lane = threadIdx.x & 31;
    const int wave = threadIdx.x >> 5;
    const int r    = lane & 15;
    const int hi   = lane >> 4;

    const int nBase = blockIdx.y * 64 + (wave >> 1) * 32;
    const int bBase = blockIdx.x * 64 + (wave & 1) * 32;

    union Frag { uint4 u[2]; v16bf v; };

    v8f acc[2][2];
    #pragma unroll
    for (int i = 0; i < 2; ++i)
        #pragma unroll
        for (int j = 0; j < 2; ++j)
            acc[i][j] = (v8f){0.f,0.f,0.f,0.f,0.f,0.f,0.f,0.f};

    for (int kk = 0; kk < D_DIM; kk += 32) {
        Frag a[2], b[2];
        // A fragments: cat rows
        #pragma unroll
        for (int ti = 0; ti < 2; ++ti) {
            const unsigned short* row =
                catBF + (size_t)(nBase + ti * 16 + r) * D_DIM + kk + hi * 8;
            a[ti].u[0] = *(const uint4*)(row);        // K kb..kb+7
            a[ti].u[1] = *(const uint4*)(row + 16);   // K kb+16..kb+23
        }
        // B fragments: batch_x rows (= output columns)
        #pragma unroll
        for (int tj = 0; tj < 2; ++tj) {
            const unsigned short* row =
                xBF + (size_t)(bBase + tj * 16 + r) * D_DIM + kk + hi * 8;
            b[tj].u[0] = *(const uint4*)(row);        // K kb..kb+7
            b[tj].u[1] = *(const uint4*)(row + 16);   // K kb+16..kb+23
        }
        #pragma unroll
        for (int ti = 0; ti < 2; ++ti)
            #pragma unroll
            for (int tj = 0; tj < 2; ++tj)
                acc[ti][tj] = __builtin_amdgcn_wmma_f32_16x16x32_bf16(
                    false, a[ti].v, false, b[tj].v,
                    (short)0, acc[ti][tj], false, false);
    }

    // store per C/D layout
    #pragma unroll
    for (int ti = 0; ti < 2; ++ti)
        #pragma unroll
        for (int tj = 0; tj < 2; ++tj) {
            #pragma unroll
            for (int e = 0; e < 8; ++e) {
                int row = nBase + ti * 16 + e + 8 * hi;
                int col = bBase + tj * 16 + r;
                con[(size_t)row * B_DIM + col] = acc[ti][tj][e];
            }
        }
}

// ---------------------------------------------------------------------------
// Kernel 4: per column b: s4 = sum con^4, cmax = max con
//   inv_norm[b] = 1/max(s4^0.25, 1e-12);  m[b] = cmax * inv_norm
// lane-per-column -> fully coalesced row reads
// ---------------------------------------------------------------------------
__global__ void colstats_kernel(const float* __restrict__ con,
                                float* __restrict__ inv_norm,
                                float* __restrict__ colmax) {
    const int b = blockIdx.x * blockDim.x + threadIdx.x;
    if (b >= B_DIM) return;
    float s4 = 0.f;
    float cm = -3.402823466e+38f;
    for (int n = 0; n < N_DIM; ++n) {
        float v  = con[(size_t)n * B_DIM + b];
        float v2 = v * v;
        s4 = fmaf(v2, v2, s4);
        cm = fmaxf(cm, v);
    }
    float norm = __powf(s4, 0.25f);
    float inv  = 1.0f / fmaxf(norm, 1e-12f);
    inv_norm[b] = inv;
    colmax[b]   = cm * inv;   // max of normalized column (norm > 0)
}

// ---------------------------------------------------------------------------
// Kernel 5: per column b:
//   Z = sum_n exp(con*inv - m);  num[s] = sum_{n in chunk s} y[n]*exp(...)
//   out[b] = sigmoid( sum_s num[s]/Z * theta[s,b] + bias )
// ---------------------------------------------------------------------------
__global__ void softmax_out_kernel(const float* __restrict__ con,
                                   const int* __restrict__ y,
                                   const float* __restrict__ inv_norm,
                                   const float* __restrict__ colmax,
                                   const float* __restrict__ theta,
                                   const float* __restrict__ bias,
                                   float* __restrict__ out) {
    const int b = blockIdx.x * blockDim.x + threadIdx.x;
    if (b >= B_DIM) return;
    const float inv = inv_norm[b];
    const float m   = colmax[b];
    float z = 0.f;
    float num[S_DIM];
    #pragma unroll
    for (int s = 0; s < S_DIM; ++s) {
        float ns = 0.f;
        const int n0 = s * NCH;
        for (int n = n0; n < n0 + NCH; ++n) {
            float e = __expf(fmaf(con[(size_t)n * B_DIM + b], inv, -m));
            z += e;
            float yn = (float)y[n];   // uniform across lanes -> scalar load
            ns = fmaf(yn, e, ns);
        }
        num[s] = ns;
    }
    float acc = 0.f;
    const float invz = 1.0f / z;
    #pragma unroll
    for (int s = 0; s < S_DIM; ++s)
        acc = fmaf(num[s] * invz, theta[(size_t)s * B_DIM + b], acc);
    acc += bias[0];
    out[b] = 1.0f / (1.0f + __expf(-acc));
}

// ---------------------------------------------------------------------------
// Launch
// ---------------------------------------------------------------------------
extern "C" void kernel_launch(void* const* d_in, const int* in_sizes, int n_in,
                              void* d_out, int out_size, void* d_ws, size_t ws_size,
                              hipStream_t stream) {
    const float* batch_x = (const float*)d_in[0];   // [B, D]
    const float* cat     = (const float*)d_in[1];   // [N, D]
    const int*   y       = (const int*)d_in[2];     // [N]
    const float* phi     = (const float*)d_in[3];   // [S, D]
    const float* bias    = (const float*)d_in[4];   // [1]
    float* out = (float*)d_out;                     // [B]

    // workspace layout (all chunk sizes are 256B-aligned multiples)
    char* ws = (char*)d_ws;
    size_t off = 0;
    float* conF = (float*)(ws + off);           off += (size_t)N_DIM * B_DIM * 4;  // 134 MB
    unsigned short* catBF = (unsigned short*)(ws + off); off += (size_t)N_DIM * D_DIM * 2;
    unsigned short* xBF   = (unsigned short*)(ws + off); off += (size_t)B_DIM * D_DIM * 2;
    float* theta    = (float*)(ws + off);       off += (size_t)S_DIM * B_DIM * 4;
    float* inv_norm = (float*)(ws + off);       off += (size_t)B_DIM * 4;
    float* colmax   = (float*)(ws + off);       off += (size_t)B_DIM * 4;
    (void)ws_size; (void)n_in; (void)in_sizes; (void)out_size;

    // 1) bf16 conversions
    {
        int nx = B_DIM * D_DIM;
        cvt_bf16_kernel<<<(nx + 1023) / 1024, 256, 0, stream>>>(batch_x, xBF, nx);
        int nc = N_DIM * D_DIM;
        cvt_bf16_kernel<<<(nc + 1023) / 1024, 256, 0, stream>>>(cat, catBF, nc);
    }

    // 2) theta (8 waves / block of 256 -> 8 b's per block)
    theta_kernel<<<B_DIM / 8, 256, 0, stream>>>(batch_x, phi, theta);

    // 3) WMMA GEMM con[N,B]
    {
        dim3 grid(B_DIM / 64, N_DIM / 64);
        gemm_con_kernel<<<grid, 128, 0, stream>>>(catBF, xBF, conF);
    }

    // 4) column stats (4-norm + max)
    colstats_kernel<<<B_DIM / 256, 256, 0, stream>>>(conF, inv_norm, colmax);

    // 5) softmax + weighted sum + sigmoid
    softmax_out_kernel<<<B_DIM / 256, 256, 0, stream>>>(conF, y, inv_norm, colmax,
                                                        theta, bias, out);
}
